// MultiHeadCrossAttention_82377472737824
// MI455X (gfx1250) — matrix operand order repaired and checked by reading, not compiled, not run on hardware
//
#include <hip/hip_runtime.h>
#include <hip/hip_bf16.h>

// MI455X / gfx1250: wave32, WMMA 16x16x32 f16 -> f32 accum, TDM async staging.
typedef __attribute__((ext_vector_type(16))) _Float16 v16h;
typedef __attribute__((ext_vector_type(8)))  _Float16 v8h;
typedef __attribute__((ext_vector_type(8)))  float    v8f;
typedef __attribute__((ext_vector_type(4)))  unsigned int u4v;
typedef __attribute__((ext_vector_type(8)))  int          i8v;
typedef __attribute__((ext_vector_type(4)))  int          i4v;

#define BB  8
#define CC  512
#define LL  1024
#define NHH 8
#define HD  64
#define MTOT (BB * LL)          // 8192 rows total
#define WELEM (CC * CC)         // 262144 per weight matrix

union AF { v16h v; v8h h[2]; };

static __device__ __forceinline__ v8f wmma32(v16h a, v16h b, v8f c) {
  // 8 args: (neg_a, A, neg_b, B, c_mod, C, reuse_a, reuse_b)
  return __builtin_amdgcn_wmma_f32_16x16x32_f16(false, a, false, b, (short)0, c,
                                                false, false);
}

// ---------------------------------------------------------------------------
// Tensor Data Mover: 2-D tile load Global -> LDS.
// Descriptor bitfields per CDNA5 ISA ch.10.8 (D# group0/group1).
// dims/strides in data_size units (here data_size = 2 bytes, f16).
// This toolchain exposes the 6-arg builtin: (g0, g1, g2, g3, g4, cpol).
// ---------------------------------------------------------------------------
static __device__ __forceinline__ void tdm_load_2d(
    const void* gptr, void* lptr,
    unsigned tile_d0, unsigned tile_d1, unsigned stride0) {
  unsigned long long ga = (unsigned long long)(uintptr_t)gptr;
  // generic LDS pointer: addr[31:0] is the LDS byte offset (aperture in MSBs)
  unsigned lds = (unsigned)(uintptr_t)lptr;
  u4v g0;
  g0.x = 1u;                                            // count=1, no gather
  g0.y = lds;                                           // lds_addr
  g0.z = (unsigned)ga;                                  // global_addr[31:0]
  g0.w = (unsigned)((ga >> 32) & 0x1FFFFFFu) | (2u << 30); // addr[56:32]|type=2
  i8v g1;
  g1[0] = 0x10000;                                      // data_size=1 (2B)
  g1[1] = (int)((tile_d0 & 0xFFFFu) << 16);             // tensor_dim0 = tile_d0
  g1[2] = (int)((tile_d0 >> 16) | ((tile_d1 & 0xFFFFu) << 16)); // tensor_dim1
  g1[3] = (int)((tile_d1 >> 16) | (tile_d0 << 16));     // tile_dim0
  g1[4] = (int)(tile_d1 & 0xFFFFu);                     // tile_dim1, tile_dim2=0
  g1[5] = (int)stride0;                                 // tensor_dim0_stride lo
  g1[6] = 0;                                            // stride0 hi, stride1 lo
  g1[7] = 0;
  i4v z4 = {0, 0, 0, 0};
  i8v z8 = {0, 0, 0, 0, 0, 0, 0, 0};
  __builtin_amdgcn_tensor_load_to_lds(g0, g1, z4, z4, z8, 0);
}

// ---------------------------------------------------------------------------
// Pack inputs: [B,C,H,W] f32 -> [B, L, C] f16 (L = H*W), for query and key.
// ---------------------------------------------------------------------------
__global__ __launch_bounds__(256) void pack_x_kernel(
    const float* __restrict__ q, const float* __restrict__ k,
    _Float16* __restrict__ xq, _Float16* __restrict__ xk) {
  int idx = blockIdx.x * 256 + threadIdx.x;      // over B*L*C
  int c = idx & (CC - 1);
  int l = (idx >> 9) & (LL - 1);
  int b = idx >> 19;
  int in_idx = (b * CC + c) * LL + l;
  xq[idx] = (_Float16)q[in_idx];
  xk[idx] = (_Float16)k[in_idx];
}

// ---------------------------------------------------------------------------
// Pack weights: f32 -> f16, row-major [j][c] (B-matrix column n == W row n).
// ---------------------------------------------------------------------------
__global__ __launch_bounds__(256) void pack_w_kernel(
    const float* __restrict__ wq, const float* __restrict__ wk,
    const float* __restrict__ wv, const float* __restrict__ wo,
    _Float16* __restrict__ w16) {
  int i = blockIdx.x * 256 + threadIdx.x;        // over 512*512
  w16[i]             = (_Float16)wq[i];
  w16[WELEM + i]     = (_Float16)wk[i];
  w16[2 * WELEM + i] = (_Float16)wv[i];
  w16[3 * WELEM + i] = (_Float16)wo[i];
}

// ---------------------------------------------------------------------------
// GEMM: out[m][n] = sum_c X[m][c] * W[n][c] + bias[n]   (M=8192, N=K=512)
// Workgroup = 8 waves; wave tile 16(M) x 64(N); WG tile 128 x 64.
// Weight slab (64 rows x 512) is DMA'd once into LDS by the TDM and shared by
// all 8 waves; k-loop B-fragments come from LDS (ds_load_b128).
// mode 0: f16 out, row-major [m][n], value = (acc+bias)*scale   (Q, K proj)
// mode 1: f16 out, transposed  [b][n][l]                        (V -> V^T)
// mode 2: f32 out, transposed [b][n][l] + residual(query+key)   (O -> d_out)
// ---------------------------------------------------------------------------
__global__ __launch_bounds__(256) void proj_kernel(
    const _Float16* __restrict__ X, const _Float16* __restrict__ W,
    const float* __restrict__ bias,
    _Float16* __restrict__ out16, float* __restrict__ out32,
    const float* __restrict__ resq, const float* __restrict__ resk,
    float scale, int mode) {
  __shared__ __align__(16) _Float16 wlds[64 * CC];     // 64 KB weight slab

  const int lane  = threadIdx.x & 31;
  const int wv    = threadIdx.x >> 5;
  const int ln15  = lane & 15;
  const int hl    = lane >> 4;
  const int mBase = blockIdx.x * 128 + wv * 16;
  const int nBase = blockIdx.y * 64;

  if (wv == 0) {
    // whole 64x512 f16 slab, rows contiguous in LDS
    tdm_load_2d(W + nBase * CC, wlds, CC, 64, CC);
    __builtin_amdgcn_s_wait_tensorcnt((short)0);
  }
  __syncthreads();

  const _Float16* xrow = X + (mBase + ln15) * CC;      // A: row = lane%16

  v8f acc[4];
  acc[0] = {}; acc[1] = {}; acc[2] = {}; acc[3] = {};

  for (int kb = 0; kb < CC; kb += 32) {
    AF a;  // A-layout: lanes0-15 K{0..15}, lanes16-31 K{16..31}
    a.h[0] = *(const v8h*)(xrow + kb + hl * 16);
    a.h[1] = *(const v8h*)(xrow + kb + 8 + hl * 16);
    const _Float16* wp = wlds + ln15 * CC + kb + hl * 16; // B: col = lane%16
#pragma unroll
    for (int t = 0; t < 4; ++t)
      acc[t] = wmma32(a.v, *(const v16h*)(wp + t * 16 * CC), acc[t]);
  }

#pragma unroll
  for (int t = 0; t < 4; ++t) {
    const int n = nBase + t * 16 + ln15;               // D: col = lane%16
    const float bval = bias[n];
#pragma unroll
    for (int v = 0; v < 8; ++v) {
      const int m = mBase + v + hl * 8;                // D: row = vgpr+8*(lane/16)
      const float val = (acc[t][v] + bval) * scale;
      if (mode == 0) {
        out16[m * CC + n] = (_Float16)val;
      } else {
        const int b = m >> 10, l = m & (LL - 1);
        const int idx = (b * CC + n) * LL + l;         // [b][j][l]
        if (mode == 1) out16[idx] = (_Float16)val;
        else           out32[idx] = val + resq[idx] + resk[idx];
      }
    }
  }
}

// ---------------------------------------------------------------------------
// Flash attention per (b, h): Q [l][d] f16, K [l][d] f16, V^T [d][l] f16.
// Workgroup: 8 waves x 16 queries = 128 queries; 32-key blocks.
// K/V tiles are double-buffered in LDS via TDM (wave 0 issues, TENSORcnt
// tracks completion; TDM ops retire in-order so wait<=2 releases the current
// pair while the next pair is still in flight).
// ---------------------------------------------------------------------------
__global__ __launch_bounds__(256) void attn_kernel(
    const _Float16* __restrict__ Qh,   // [B*L][512] (pre-scaled by hd^-0.5)
    const _Float16* __restrict__ Kh,   // [B*L][512]
    const _Float16* __restrict__ Vt,   // [B*512][1024]
    _Float16* __restrict__ O) {        // [B*L][512]
  __shared__ __align__(16) _Float16 kbuf[2][32 * HD];  // 2 x 4KB: [key][d]
  __shared__ __align__(16) _Float16 vbuf[2][HD * 32];  // 2 x 4KB: [d][key]
  __shared__ __align__(16) _Float16 pbuf[8][16 * 32];  // per-wave P staging

  const int lane  = threadIdx.x & 31;
  const int wv    = threadIdx.x >> 5;
  const int ln15  = lane & 15;
  const int hl    = lane >> 4;
  const int bh    = blockIdx.x;
  const int b     = bh >> 3, h = bh & 7;
  const int qbase = blockIdx.y * 128 + wv * 16;

  // Q A-fragments for d = 0..31 and d = 32..63
  const _Float16* qrow = Qh + (b * LL + qbase + ln15) * CC + h * HD;
  AF aq0, aq1;
  aq0.h[0] = *(const v8h*)(qrow + hl * 16);
  aq0.h[1] = *(const v8h*)(qrow + 8 + hl * 16);
  aq1.h[0] = *(const v8h*)(qrow + 32 + hl * 16);
  aq1.h[1] = *(const v8h*)(qrow + 40 + hl * 16);

  const _Float16* kbarr = Kh + (b * LL) * CC + h * HD;    // + key*CC
  const _Float16* vbarr = Vt + (b * CC + h * HD) * LL;    // + d*LL + key

  float mrow[8], lrow[8];
  v8f oacc[4];
#pragma unroll
  for (int v = 0; v < 8; ++v) { mrow[v] = -3.0e38f; lrow[v] = 0.0f; }
  oacc[0] = {}; oacc[1] = {}; oacc[2] = {}; oacc[3] = {};

  _Float16* pw = &pbuf[wv][0];
  const v8f zc = {};
  const int NIT = LL / 32;                 // 32 iterations

  // Prologue: DMA first K/V tile pair into buffer 0
  if (wv == 0) {
    tdm_load_2d(kbarr, kbuf[0], HD, 32, CC);   // 32 key rows x 64 d
    tdm_load_2d(vbarr, vbuf[0], 32, HD, LL);   // 64 d rows x 32 keys
  }

  for (int it = 0; it < NIT; ++it) {
    const int kb = it * 32;
    if (wv == 0) {
      if (it + 1 < NIT) {
        const int kn = kb + 32;
        tdm_load_2d(kbarr + kn * CC, kbuf[(it + 1) & 1], HD, 32, CC);
        tdm_load_2d(vbarr + kn,      vbuf[(it + 1) & 1], 32, HD, LL);
        __builtin_amdgcn_s_wait_tensorcnt((short)2);  // current pair done
      } else {
        __builtin_amdgcn_s_wait_tensorcnt((short)0);
      }
    }
    __syncthreads();                       // current tiles visible to all waves

    const _Float16* kt = kbuf[it & 1];
    const _Float16* vt = vbuf[it & 1];

    // S = Q K^T for two 16-key tiles (K-dim = d = 64 -> 2 WMMA steps each)
    const _Float16* k0 = kt + ln15 * HD + hl * 16;
    const _Float16* k1 = kt + (ln15 + 16) * HD + hl * 16;
    v8f s0 = wmma32(aq0.v, *(const v16h*)(k0),      zc);
    s0     = wmma32(aq1.v, *(const v16h*)(k0 + 32), s0);
    v8f s1 = wmma32(aq0.v, *(const v16h*)(k1),      zc);
    s1     = wmma32(aq1.v, *(const v16h*)(k1 + 32), s1);

    // Online softmax: per-row (across the 16-lane half of the wave)
#pragma unroll
    for (int v = 0; v < 8; ++v) {
      float tmax = fmaxf(s0[v], s1[v]);
#pragma unroll
      for (int msk = 1; msk < 16; msk <<= 1)
        tmax = fmaxf(tmax, __shfl_xor(tmax, msk, 32));
      const float mnew = fmaxf(mrow[v], tmax);
      const float corr = __expf(mrow[v] - mnew);
      const float p0 = __expf(s0[v] - mnew);
      const float p1 = __expf(s1[v] - mnew);
      float rs = p0 + p1;
#pragma unroll
      for (int msk = 1; msk < 16; msk <<= 1)
        rs += __shfl_xor(rs, msk, 32);
      lrow[v] = lrow[v] * corr + rs;
      mrow[v] = mnew;
#pragma unroll
      for (int t = 0; t < 4; ++t) oacc[t][v] *= corr;
      const int r = v + hl * 8;                 // D-layout row
      pw[r * 32 + ln15]      = (_Float16)p0;    // row-major 16x32 P tile
      pw[r * 32 + 16 + ln15] = (_Float16)p1;
    }

    // Re-read P as an A-fragment (wave-synchronous; DS ops in-order per wave)
    AF pa;
    pa.h[0] = *(const v8h*)(pw + ln15 * 32 + hl * 16);
    pa.h[1] = *(const v8h*)(pw + ln15 * 32 + 8 + hl * 16);

    // O += P @ V  (4 d-tiles of 16)
#pragma unroll
    for (int t = 0; t < 4; ++t) {
      const _Float16* vp = vt + (t * 16 + ln15) * 32 + hl * 16;
      oacc[t] = wmma32(pa.v, *(const v16h*)(vp), oacc[t]);
    }
    __syncthreads();    // all waves done before this buffer is overwritten
  }

  // Normalize and store O as [b][l][c] f16
#pragma unroll
  for (int t = 0; t < 4; ++t) {
    const int cg = h * HD + t * 16 + ln15;
#pragma unroll
    for (int v = 0; v < 8; ++v) {
      const int l = qbase + v + hl * 8;
      O[(b * LL + l) * CC + cg] = (_Float16)(oacc[t][v] / lrow[v]);
    }
  }
}

// ---------------------------------------------------------------------------
extern "C" void kernel_launch(void* const* d_in, const int* in_sizes, int n_in,
                              void* d_out, int out_size, void* d_ws, size_t ws_size,
                              hipStream_t stream) {
  const float* q  = (const float*)d_in[0];
  const float* k  = (const float*)d_in[1];
  const float* Wq = (const float*)d_in[2];
  const float* bq = (const float*)d_in[3];
  const float* Wk = (const float*)d_in[4];
  const float* bk = (const float*)d_in[5];
  const float* Wv = (const float*)d_in[6];
  const float* bv = (const float*)d_in[7];
  const float* Wo = (const float*)d_in[8];
  const float* bo = (const float*)d_in[9];
  float* out = (float*)d_out;

  // Workspace layout (all f16, ~52 MB total, fits the 192 MB L2)
  _Float16* Xq   = (_Float16*)d_ws;               // [8192][512]  qf
  _Float16* Xk   = Xq + (size_t)MTOT * CC;        // [8192][512]  kf
  _Float16* W16  = Xk + (size_t)MTOT * CC;        // 4 x [512][512]
  _Float16* Qh   = W16 + 4 * (size_t)WELEM;       // [8192][512]  q * hd^-0.5
  _Float16* Kh   = Qh + (size_t)MTOT * CC;        // [8192][512]
  _Float16* Vt   = Kh + (size_t)MTOT * CC;        // [B*512][1024] V^T
  _Float16* Of   = Vt + (size_t)MTOT * CC;        // [8192][512]  attn out

  dim3 blk(256);
  pack_x_kernel<<<(MTOT * CC) / 256, blk, 0, stream>>>(q, k, Xq, Xk);
  pack_w_kernel<<<WELEM / 256, blk, 0, stream>>>(Wq, Wk, Wv, Wo, W16);

  dim3 pg(MTOT / 128, CC / 64);   // 64 x 8
  const float scaleq = 0.125f;    // hd^-0.5, folded into Q projection
  proj_kernel<<<pg, blk, 0, stream>>>(Xq, W16,             bq, Qh, nullptr,
                                      nullptr, nullptr, scaleq, 0);
  proj_kernel<<<pg, blk, 0, stream>>>(Xk, W16 + WELEM,     bk, Kh, nullptr,
                                      nullptr, nullptr, 1.0f, 0);
  proj_kernel<<<pg, blk, 0, stream>>>(Xk, W16 + 2 * WELEM, bv, Vt, nullptr,
                                      nullptr, nullptr, 1.0f, 1);

  dim3 ag(BB * NHH, LL / 128);    // 64 x 8
  attn_kernel<<<ag, blk, 0, stream>>>(Qh, Kh, Vt, Of);

  // O projection + bias + residual(qf+kf) + writeback to NCHW f32
  proj_kernel<<<pg, blk, 0, stream>>>(Of, W16 + 3 * WELEM, bo, nullptr, out,
                                      q, k, 1.0f, 2);
}
